// MHA_GQA_40879498728814
// MI455X (gfx1250) — matrix operand order repaired and checked by reading, not compiled, not run on hardware
//
#include <hip/hip_runtime.h>
#include <hip/hip_bf16.h>
#include <math.h>

typedef _Float16 h16;
typedef __attribute__((ext_vector_type(16))) _Float16 v16h;
typedef __attribute__((ext_vector_type(8)))  _Float16 v8h;
typedef __attribute__((ext_vector_type(8)))  float    v8f;

union V16 { v16h v; v8h h[2]; };

__device__ __forceinline__ v8f wmma_f16(v16h a, v16h b, v8f c) {
    return __builtin_amdgcn_wmma_f32_16x16x32_f16(
        /*neg_a=*/false, a, /*neg_b=*/false, b,
        /*c_mod=*/(short)0, c, /*reuse_a=*/false, /*reuse_b=*/false);
}

// ---------------- problem constants ----------------
constexpr int B_  = 2;
constexpr int T_  = 2048;
constexpr int C_  = 2048;
constexpr int H_  = 32;
constexpr int KH_ = 8;          // kv heads
constexpr int HD_ = 64;         // head dim
constexpr int NQKV_ = (H_ + 2 * KH_) * HD_;   // 3072

// ---------------- fp32 -> fp16 convert ----------------
__global__ void cvt_f32_f16_kernel(const float* __restrict__ src,
                                   h16* __restrict__ dst, int n) {
    int i = blockIdx.x * blockDim.x + threadIdx.x;
    if (i < n) dst[i] = (h16)src[i];
}

// ---------------- GEMM: C[M,N] = A[M,Kc] * W[N,Kc]^T (row-major, f16 in, f32 out)
// WG tile 64(M) x 256(N), 8 waves of 32x64 (2x4 WMMA tiles), K step 32,
// LDS double-buffered: one barrier per K-step, 8 WMMAs per 12 ds_load_b128 per wave.
__launch_bounds__(256)
__global__ void gemm_nt_kernel(const h16* __restrict__ A, const h16* __restrict__ W,
                               float* __restrict__ C, int M, int N, int Kc) {
    constexpr int LDT = 40;                 // padded LDS row stride (halves), 80B
    __shared__ h16 As[2][64 * LDT];         // 2 x  5,120 B
    __shared__ h16 Ws[2][256 * LDT];        // 2 x 20,480 B

    const int t    = threadIdx.x;
    const int wave = t >> 5;
    const int lane = t & 31;
    const int g    = lane >> 4;             // lane group (0/1)
    const int ln   = lane & 15;
    const int m0   = blockIdx.y * 64;
    const int n0   = blockIdx.x * 256;
    const int wm   = (wave & 1) * 32;       // wave M offset inside WG tile
    const int wn   = (wave >> 1) * 64;      // wave N offset

    const int arow = t >> 2, akoff = (t & 3) * 8;    // A: 64 rows, 4 threads/row
    const h16* arowp = &A[(size_t)(m0 + arow) * Kc + akoff];
    const h16* wrowp = &W[(size_t)(n0 + t) * Kc];    // W: 256 rows, 1 thread/row

    // prologue: stage tile 0 into buffer 0
    *(v8h*)&As[0][arow * LDT + akoff] = *(const v8h*)(arowp);
#pragma unroll
    for (int u = 0; u < 4; ++u)
        *(v8h*)&Ws[0][t * LDT + u * 8] = *(const v8h*)(wrowp + u * 8);
    __syncthreads();

    v8f acc[2][4] = {};

    for (int kb = 0; kb < Kc; kb += 32) {
        const int  buf  = (kb >> 5) & 1;
        const bool more = (kb + 32) < Kc;

        // issue next-tile global loads early (overlap with WMMA block)
        v8h anext;
        v8h wnext[4];
        if (more) {
            anext = *(const v8h*)(arowp + kb + 32);
#pragma unroll
            for (int u = 0; u < 4; ++u)
                wnext[u] = *(const v8h*)(wrowp + kb + 32 + u * 8);
        }

        V16 af[2], bf[4];
#pragma unroll
        for (int i = 0; i < 2; ++i) {
            const int m = wm + i * 16 + ln;
            af[i].h[0] = *(const v8h*)&As[buf][m * LDT + g * 8];
            af[i].h[1] = *(const v8h*)&As[buf][m * LDT + 16 + g * 8];
        }
#pragma unroll
        for (int j = 0; j < 4; ++j) {
            const int n = wn + j * 16 + ln;
            bf[j].h[0] = *(const v8h*)&Ws[buf][n * LDT + g * 8];
            bf[j].h[1] = *(const v8h*)&Ws[buf][n * LDT + 16 + g * 8];
        }
#pragma unroll
        for (int i = 0; i < 2; ++i)
#pragma unroll
            for (int j = 0; j < 4; ++j)
                acc[i][j] = wmma_f16(af[i].v, bf[j].v, acc[i][j]);

        if (more) {
            *(v8h*)&As[buf ^ 1][arow * LDT + akoff] = anext;
#pragma unroll
            for (int u = 0; u < 4; ++u)
                *(v8h*)&Ws[buf ^ 1][t * LDT + u * 8] = wnext[u];
            __syncthreads();
        }
    }

#pragma unroll
    for (int i = 0; i < 2; ++i)
#pragma unroll
        for (int j = 0; j < 4; ++j)
#pragma unroll
            for (int r = 0; r < 8; ++r) {
                const int row = m0 + wm + i * 16 + g * 8 + r;
                const int col = n0 + wn + j * 16 + ln;
                C[(size_t)row * N + col] = acc[i][j][r];
            }
}

// ---------------- RoPE + split/pack to head-major f16 q/k/v ----------------
__global__ void rope_pack_kernel(const float* __restrict__ qkv,
                                 h16* __restrict__ qh, h16* __restrict__ kh,
                                 h16* __restrict__ vh) {
    const int idx = blockIdx.x * blockDim.x + threadIdx.x;   // over (B*T) * 3072
    const int c = idx % NQKV_;
    const int m = idx / NQKV_;
    if (m >= B_ * T_) return;
    const int b = m / T_, tt = m % T_;
    const float* row = qkv + (size_t)m * NQKV_;

    if (c < H_ * HD_) {                       // Q, with rotary
        const int h = c / HD_, d = c % HD_, i = d & 31;
        const float ang = (float)tt * __powf(10000.f, -(float)i / 32.f);
        float sn, cs; __sincosf(ang, &sn, &cs);
        const float t1 = row[h * HD_ + i];
        const float t2 = row[h * HD_ + 32 + i];
        const float v = (d < 32) ? (t1 * cs - t2 * sn) : (t2 * cs + t1 * sn);
        qh[(((size_t)b * H_ + h) * T_ + tt) * HD_ + d] = (h16)v;
    } else if (c < (H_ + KH_) * HD_) {        // K, with rotary
        const int cc = c - H_ * HD_;
        const int h = cc / HD_, d = cc % HD_, i = d & 31;
        const int base = H_ * HD_ + h * HD_;
        const float ang = (float)tt * __powf(10000.f, -(float)i / 32.f);
        float sn, cs; __sincosf(ang, &sn, &cs);
        const float t1 = row[base + i];
        const float t2 = row[base + 32 + i];
        const float v = (d < 32) ? (t1 * cs - t2 * sn) : (t2 * cs + t1 * sn);
        kh[(((size_t)b * KH_ + h) * T_ + tt) * HD_ + d] = (h16)v;
    } else {                                  // V passthrough
        const int cc = c - (H_ + KH_) * HD_;
        const int h = cc / HD_, d = cc % HD_;
        vh[(((size_t)b * KH_ + h) * T_ + tt) * HD_ + d] =
            (h16)row[(H_ + KH_) * HD_ + h * HD_ + d];
    }
}

// ---------------- causal flash attention (GQA), WMMA inner products ----------------
// grid: (T/128, B*H); block: 256 (8 waves). Wave owns 16 q rows.
__launch_bounds__(256)
__global__ void attn_kernel(const h16* __restrict__ qh, const h16* __restrict__ kh,
                            const h16* __restrict__ vh, h16* __restrict__ oh) {
    constexpr int QT = 128;     // q rows per WG
    constexpr int KT = 32;      // keys per iteration

    __shared__ h16 Ks[KT][HD_];        // K tile, row-major  (32 x 64)
    __shared__ h16 Vt[HD_][KT];        // V tile, transposed (64 x 32)
    __shared__ h16 Ps[8][16][KT];      // per-wave P tile 16 x 32 (D->A transpose bounce)

    const int t    = threadIdx.x;
    const int wave = t >> 5;
    const int lane = t & 31;
    const int g    = lane >> 4;
    const int ln   = lane & 15;

    const int bh  = blockIdx.y;
    const int b   = bh / H_, h = bh % H_;
    const int khd = h / (H_ / KH_);           // kv head (rep = 4)
    const int q0  = blockIdx.x * QT;
    const int qb  = q0 + wave * 16;

    const h16* qbase = qh + (((size_t)b * H_ + h) * T_) * HD_;
    const h16* kbase = kh + (((size_t)b * KH_ + khd) * T_) * HD_;
    const h16* vbase = vh + (((size_t)b * KH_ + khd) * T_) * HD_;

    // Q A-frags for both 32-wide K-dim chunks (reused across all key tiles)
    V16 qf[2];
#pragma unroll
    for (int cch = 0; cch < 2; ++cch) {
        const h16* qrow = qbase + (size_t)(qb + ln) * HD_ + cch * 32;
        qf[cch].h[0] = *(const v8h*)(qrow + g * 8);
        qf[cch].h[1] = *(const v8h*)(qrow + 16 + g * 8);
    }

    v8f oacc[4] = {};                     // O: 16 q rows x 64 dims (4 N-chunks)
    float mrow[8], lrow[8];
#pragma unroll
    for (int r = 0; r < 8; ++r) { mrow[r] = -1e30f; lrow[r] = 0.f; }

    const int kend = q0 + QT;             // WG-uniform causal bound
    for (int kt = 0; kt < kend; kt += KT) {
        __syncthreads();
        {   // cooperative stage: K row-major, V transposed
            const int row = t >> 3, off = (t & 7) * 8;
            *(v8h*)&Ks[row][off] = *(const v8h*)&kbase[(size_t)(kt + row) * HD_ + off];
            v8h vv = *(const v8h*)&vbase[(size_t)(kt + row) * HD_ + off];
#pragma unroll
            for (int e = 0; e < 8; ++e) Vt[off + e][row] = vv[e];
        }
        __syncthreads();

        // S = q . k^T : two 16-key subtiles, K-dim 64 = 2 chained WMMAs each
        v8f sacc[2] = {};
#pragma unroll
        for (int ns = 0; ns < 2; ++ns)
#pragma unroll
            for (int cch = 0; cch < 2; ++cch) {
                V16 bf;
                const h16* krow = &Ks[ns * 16 + ln][cch * 32 + g * 8];
                bf.h[0] = *(const v8h*)(krow);
                bf.h[1] = *(const v8h*)(krow + 16);
                sacc[ns] = wmma_f16(qf[cch].v, bf.v, sacc[ns]);
            }

        // online softmax per q row (rows live in 16-lane halves)
        float p[2][8];
#pragma unroll
        for (int r = 0; r < 8; ++r) {
            const int qr = qb + g * 8 + r;
            float mx = -1e30f;
#pragma unroll
            for (int ns = 0; ns < 2; ++ns) {
                const int kc = kt + ns * 16 + ln;
                float s = sacc[ns][r] * 0.125f;          // 1/sqrt(64)
                s = (kc <= qr) ? s : -1e30f;
                p[ns][r] = s;
                mx = fmaxf(mx, s);
            }
            for (int o = 1; o < 16; o <<= 1) mx = fmaxf(mx, __shfl_xor(mx, o, 32));
            const float mnew  = fmaxf(mrow[r], mx);
            const float alpha = __expf(mrow[r] - mnew);
            float rs = 0.f;
#pragma unroll
            for (int ns = 0; ns < 2; ++ns) {
                const int kc = kt + ns * 16 + ln;
                const float e = (kc <= qr) ? __expf(p[ns][r] - mnew) : 0.f;
                p[ns][r] = e;
                rs += e;
            }
            for (int o = 1; o < 16; o <<= 1) rs += __shfl_xor(rs, o, 32);
            lrow[r] = lrow[r] * alpha + rs;
            mrow[r] = mnew;
#pragma unroll
            for (int nc = 0; nc < 4; ++nc) oacc[nc][r] *= alpha;
        }

        // P: D-layout -> A-layout via per-wave LDS bounce (DS ops in-order per wave)
#pragma unroll
        for (int r = 0; r < 8; ++r) {
            Ps[wave][g * 8 + r][ln]      = (h16)p[0][r];
            Ps[wave][g * 8 + r][16 + ln] = (h16)p[1][r];
        }
        asm volatile("s_wait_dscnt 0" ::: "memory");
        V16 pf;
        pf.h[0] = *(const v8h*)&Ps[wave][ln][g * 8];
        pf.h[1] = *(const v8h*)&Ps[wave][ln][16 + g * 8];

        // O += P . V : 4 WMMAs over the 64 output dims
#pragma unroll
        for (int nc = 0; nc < 4; ++nc) {
            V16 bf;
            const h16* vrow = &Vt[nc * 16 + ln][g * 8];
            bf.h[0] = *(const v8h*)(vrow);
            bf.h[1] = *(const v8h*)(vrow + 16);
            oacc[nc] = wmma_f16(pf.v, bf.v, oacc[nc]);
        }
    }

    // epilogue: divide by l, scatter f16 to [B*T, C] layout for the Wo GEMM
#pragma unroll
    for (int nc = 0; nc < 4; ++nc)
#pragma unroll
        for (int r = 0; r < 8; ++r) {
            const int qr = qb + g * 8 + r;
            const float o = oacc[nc][r] / lrow[r];
            oh[((size_t)(b * T_ + qr)) * C_ + h * HD_ + nc * 16 + ln] = (h16)o;
        }
}

// ---------------- host orchestration ----------------
extern "C" void kernel_launch(void* const* d_in, const int* in_sizes, int n_in,
                              void* d_out, int out_size, void* d_ws, size_t ws_size,
                              hipStream_t stream) {
    const float* x    = (const float*)d_in[0];
    const float* Wqkv = (const float*)d_in[1];
    const float* Wo   = (const float*)d_in[2];
    float* out = (float*)d_out;
    char* ws = (char*)d_ws;

    const int M = B_ * T_;                      // 4096
    // workspace layout (bytes); oh aliases qkvf (qkvf dead after rope_pack)
    h16*   xh    = (h16*)(ws);                          // 16,777,216
    h16*   wqkvh = (h16*)(ws + 16777216);               // 12,582,912
    h16*   woh   = (h16*)(ws + 29360128);               //  8,388,608
    float* qkvf  = (float*)(ws + 37748736);             // 50,331,648
    h16*   oh    = (h16*)(ws + 37748736);               // 16,777,216 (alias, used later)
    h16*   qh    = (h16*)(ws + 88080384);               // 16,777,216
    h16*   kh    = (h16*)(ws + 104857600);              //  4,194,304
    h16*   vh    = (h16*)(ws + 109051904);              //  4,194,304

    const int nx = M * C_, nwq = NQKV_ * C_, nwo = C_ * C_;
    cvt_f32_f16_kernel<<<(nx  + 255) / 256, 256, 0, stream>>>(x,    xh,    nx);
    cvt_f32_f16_kernel<<<(nwq + 255) / 256, 256, 0, stream>>>(Wqkv, wqkvh, nwq);
    cvt_f32_f16_kernel<<<(nwo + 255) / 256, 256, 0, stream>>>(Wo,   woh,   nwo);

    // qkv = x @ Wqkv^T
    gemm_nt_kernel<<<dim3(NQKV_ / 256, M / 64), 256, 0, stream>>>(
        xh, wqkvh, qkvf, M, NQKV_, C_);

    // RoPE + head-major pack
    rope_pack_kernel<<<(M * NQKV_) / 256, 256, 0, stream>>>(qkvf, qh, kh, vh);

    // causal GQA flash attention
    attn_kernel<<<dim3(T_ / 128, B_ * H_), 256, 0, stream>>>(qh, kh, vh, oh);

    // out = attn @ Wo^T
    gemm_nt_kernel<<<dim3(C_ / 256, M / 64), 256, 0, stream>>>(
        oh, woh, out, M, C_, C_);
}
